// DGCNN_39384850104582
// MI455X (gfx1250) — matrix operand order repaired and checked by reading, not compiled, not run on hardware
//
#include <hip/hip_runtime.h>
#include <hip/hip_bf16.h>
#include <math.h>

typedef __attribute__((ext_vector_type(16))) __bf16 v16bf;
typedef __attribute__((ext_vector_type(8)))  __bf16 v8bf;
typedef __attribute__((ext_vector_type(8)))  float  v8f;

#define NPTS   1024
#define KNN    20
#define NBATCH 8
#define MROWS  (NPTS * KNN)   // 20480 edge rows per batch

// ---------------------------------------------------------------------------
// kNN: one thread per point, scalar distance scan + insertion top-K (fp32 for
// index fidelity; negligible FLOPs vs the WMMA GEMMs). grid.z = batch.
// ---------------------------------------------------------------------------
__global__ void knn_kernel(const float* __restrict__ x, int* __restrict__ idx,
                           int C) {
    int i = blockIdx.x * blockDim.x + threadIdx.x;
    int b = blockIdx.z;
    if (i >= NPTS) return;
    const float* xb = x + (size_t)b * NPTS * C;
    float bd[KNN];
    int   bi[KNN];
#pragma unroll
    for (int t = 0; t < KNN; ++t) { bd[t] = 3.0e38f; bi[t] = 0; }
    for (int j = 0; j < NPTS; ++j) {
        float d2 = 0.0f;
        for (int c = 0; c < C; ++c) {
            float diff = xb[(size_t)i * C + c] - xb[(size_t)j * C + c];
            d2 += diff * diff;
        }
        if (d2 < bd[KNN - 1]) {
            int p = KNN - 1;
            while (p > 0 && bd[p - 1] > d2) {
                bd[p] = bd[p - 1]; bi[p] = bi[p - 1]; --p;
            }
            bd[p] = d2; bi[p] = j;
        }
    }
    int* ob = idx + ((size_t)b * NPTS + i) * KNN;
#pragma unroll
    for (int t = 0; t < KNN; ++t) ob[t] = bi[t];
}

// ---------------------------------------------------------------------------
// Build edge-feature matrix E[M, Kp] (bf16, zero-padded K): row r = i*K + k:
// cols [0,C) = x_i ; [C,2C) = x_j - x_i ; rest 0.
// ---------------------------------------------------------------------------
__global__ void build_edges(const float* __restrict__ x,
                            const int* __restrict__ idx,
                            __bf16* __restrict__ E, int C, int Kp) {
    size_t tid = (size_t)blockIdx.x * blockDim.x + threadIdx.x;
    size_t total = (size_t)MROWS * Kp;
    if (tid >= total) return;
    int r = (int)(tid / Kp);
    int c = (int)(tid % Kp);
    int i = r / KNN;
    int k = r % KNN;
    float val = 0.0f;
    if (c < C) {
        val = x[(size_t)i * C + c];
    } else if (c < 2 * C) {
        int j = idx[(size_t)i * KNN + k];
        int cc = c - C;
        val = x[(size_t)j * C + cc] - x[(size_t)i * C + cc];
    }
    E[tid] = (__bf16)val;
}

// ---------------------------------------------------------------------------
// Weight pack: f32 [Kin, Ncol] -> bf16 in WMMA B-fragment order, zero-padded
// to Kp rows. Packed layout: [tile_n][k_chunk(32)][lane(32)][elem(16)],
// where lane L holds col = tile_n*16 + L%16, elem j holds K = kc*32 +
// (L/16)*8 + (j%8) + (j/8)*16  (CDNA5 ISA 7.12.2 16-bit B layout). Each lane's fragment is
// then one contiguous 32-byte load in the GEMM.
// ---------------------------------------------------------------------------
__global__ void wpack(const float* __restrict__ W, __bf16* __restrict__ out,
                      int Kin, int Kp, int Ncol) {
    int tid = blockIdx.x * blockDim.x + threadIdx.x;
    int total = Kp * Ncol;
    if (tid >= total) return;
    int nkc  = Kp >> 5;
    int j    = tid & 15;
    int lane = (tid >> 4) & 31;
    int rest = tid >> 9;           // = tn * nkc + kc
    int tn   = rest / nkc;
    int kc   = rest % nkc;
    int K    = kc * 32 + (lane >> 4) * 8 + (j & 7) + (j >> 3) * 16;
    int col  = tn * 16 + (lane & 15);
    out[tid] = (K < Kin) ? (__bf16)W[(size_t)K * Ncol + col] : (__bf16)0.0f;
}

// ---------------------------------------------------------------------------
// bf16 WMMA GEMM: Out[M,NCOL] = A[M,KP] @ W + bias (opt ReLU), W pre-packed
// in fragment order. Each wave computes a 16x32 output slab (two 16x16 tiles
// sharing one A fragment). KP and NCOL are compile-time: the K loop fully
// unrolls, fragment loads are 16B/32B contiguous (global_load_b128), and all
// 16 output stores fold their row/col offsets into the 24-bit immediate
// offset of a single-clause store burst (no u64 address-add chains).
// Tile early-out is wave-uniform so EXEC is all ones at every WMMA.
// ---------------------------------------------------------------------------
template <int KP, int NCOL, bool RELU, bool OUTBF16>
__global__ void wmma_gemm(const __bf16* __restrict__ A,
                          const __bf16* __restrict__ Bp,
                          const float* __restrict__ bias,
                          void* __restrict__ Out,
                          int M) {
    constexpr int NKC      = KP / 32;
    constexpr int TILES_N2 = NCOL / 32;            // 32-col slabs
    int wave   = blockIdx.x * (blockDim.x >> 5) + (threadIdx.x >> 5);
    int ntiles = (M >> 4) * TILES_N2;
    if (wave >= ntiles) return;                    // uniform per wave
    int tm   = wave / TILES_N2;
    int tn2  = wave % TILES_N2;
    int lane = threadIdx.x & 31;
    int half = lane >> 4;
    int l16  = lane & 15;
    int mr   = (tm << 4) + l16;                    // A row held by this lane

    const __bf16* arow = A + (size_t)mr * KP + half * 8;
    const __bf16* bp0  = Bp + (((size_t)(2 * tn2)     * NKC) * 32 + lane) * 16;
    const __bf16* bp1  = Bp + (((size_t)(2 * tn2 + 1) * NKC) * 32 + lane) * 16;

    v8f acc0 = {}, acc1 = {};
#pragma unroll
    for (int kc = 0; kc < NKC; ++kc) {
        v8bf lo = *(const v8bf*)(arow + kc * 32);       // K = kc*32+8h+(0..7)
        v8bf hi = *(const v8bf*)(arow + kc * 32 + 16);  // K = kc*32+8h+16+(0..7)
        v16bf a = __builtin_shufflevector(lo, hi, 0, 1, 2, 3, 4, 5, 6, 7,
                                          8, 9, 10, 11, 12, 13, 14, 15);
        v16bf b0 = *(const v16bf*)(bp0 + (size_t)kc * 32 * 16);
        v16bf b1 = *(const v16bf*)(bp1 + (size_t)kc * 32 * 16);
        acc0 = __builtin_amdgcn_wmma_f32_16x16x32_bf16(
            false, a, false, b0, (short)0, acc0, false, false);
        acc1 = __builtin_amdgcn_wmma_f32_16x16x32_bf16(
            false, a, false, b1, (short)0, acc1, false, false);
    }

    int nc0 = (tn2 << 5) + l16;
    int m0  = (tm << 4) + half * 8;                 // first output row of lane
    float bs0 = bias[nc0];
    float bs1 = bias[nc0 + 16];

    if (OUTBF16) {
        __bf16* obase = (__bf16*)Out + (size_t)m0 * NCOL + nc0;
#pragma unroll
        for (int v = 0; v < 8; ++v) {
            float v0 = acc0[v] + bs0;
            float v1 = acc1[v] + bs1;
            if (RELU) {
                v0 = v0 > 0.0f ? v0 : 0.0f;
                v1 = v1 > 0.0f ? v1 : 0.0f;
            }
            obase[v * NCOL]      = (__bf16)v0;      // immediate offsets
            obase[v * NCOL + 16] = (__bf16)v1;
        }
    } else {
        float* obase = (float*)Out + (size_t)m0 * NCOL + nc0;
#pragma unroll
        for (int v = 0; v < 8; ++v) {
            float v0 = acc0[v] + bs0;
            float v1 = acc1[v] + bs1;
            if (RELU) {
                v0 = v0 > 0.0f ? v0 : 0.0f;
                v1 = v1 > 0.0f ? v1 : 0.0f;
            }
            obase[v * NCOL]      = v0;
            obase[v * NCOL + 16] = v1;
        }
    }
}

// ---------------------------------------------------------------------------
// Max over K neighbors: Xout[i,c] = max_k G[i*K+k, c]
// ---------------------------------------------------------------------------
__global__ void reduce_max_k(const float* __restrict__ G,
                             float* __restrict__ Xout, int C) {
    int tid = blockIdx.x * blockDim.x + threadIdx.x;
    if (tid >= NPTS * C) return;
    int i = tid / C;
    int c = tid % C;
    float v = -3.0e38f;
#pragma unroll
    for (int k = 0; k < KNN; ++k)
        v = fmaxf(v, G[((size_t)i * KNN + k) * C + c]);
    Xout[tid] = v;
}

// ---------------------------------------------------------------------------
// Global max pool over N of concat(x1,x2,x3,x4) -> xpool [8,512]
// ---------------------------------------------------------------------------
__global__ void pool_kernel(const float* __restrict__ X1,
                            const float* __restrict__ X2,
                            const float* __restrict__ X3,
                            const float* __restrict__ X4,
                            float* __restrict__ xpool) {
    int tid = blockIdx.x * blockDim.x + threadIdx.x;
    if (tid >= NBATCH * 512) return;
    int b = tid / 512;
    int c = tid % 512;
    const float* src; int Cs, cc;
    if (c < 64)       { src = X1; Cs = 64;  cc = c; }
    else if (c < 128) { src = X2; Cs = 64;  cc = c - 64; }
    else if (c < 256) { src = X3; Cs = 128; cc = c - 128; }
    else              { src = X4; Cs = 256; cc = c - 256; }
    const float* base = src + (size_t)b * NPTS * Cs;
    float v = -3.0e38f;
    for (int n = 0; n < NPTS; ++n)
        v = fmaxf(v, base[(size_t)n * Cs + cc]);
    xpool[tid] = v;
}

// ---------------------------------------------------------------------------
// Head: lin1 + batchnorm(batch of 8) + relu + lin2 + log_softmax. 1 block.
// ---------------------------------------------------------------------------
__global__ void head_kernel(const float* __restrict__ xpool,
                            const float* __restrict__ lin1_w,
                            const float* __restrict__ lin1_b,
                            const float* __restrict__ bn_g,
                            const float* __restrict__ bn_b,
                            const float* __restrict__ lin2_w,
                            const float* __restrict__ lin2_b,
                            float* __restrict__ out) {
    __shared__ float hs[NBATCH * 1024];
    __shared__ float lg[NBATCH * 40];
    int c = threadIdx.x;  // 1024 threads
    float hb[NBATCH];
#pragma unroll
    for (int b = 0; b < NBATCH; ++b) {
        float acc = lin1_b[c];
        for (int k = 0; k < 512; ++k)
            acc += xpool[b * 512 + k] * lin1_w[(size_t)k * 1024 + c];
        hb[b] = acc;
    }
    float mean = 0.0f;
#pragma unroll
    for (int b = 0; b < NBATCH; ++b) mean += hb[b];
    mean *= (1.0f / NBATCH);
    float var = 0.0f;
#pragma unroll
    for (int b = 0; b < NBATCH; ++b) { float d = hb[b] - mean; var += d * d; }
    var *= (1.0f / NBATCH);
    float inv = rsqrtf(var + 1e-5f);
#pragma unroll
    for (int b = 0; b < NBATCH; ++b) {
        float v = bn_g[c] * (hb[b] - mean) * inv + bn_b[c];
        hs[b * 1024 + c] = v > 0.0f ? v : 0.0f;
    }
    __syncthreads();
    if (c < NBATCH * 40) {
        int b = c / 40, o = c % 40;
        float acc = lin2_b[o];
        for (int k = 0; k < 1024; ++k)
            acc += hs[b * 1024 + k] * lin2_w[(size_t)k * 40 + o];
        lg[b * 40 + o] = acc;
    }
    __syncthreads();
    if (c < NBATCH) {
        float m = -3.0e38f;
        for (int o = 0; o < 40; ++o) m = fmaxf(m, lg[c * 40 + o]);
        float s = 0.0f;
        for (int o = 0; o < 40; ++o) s += expf(lg[c * 40 + o] - m);
        float ls = logf(s);
        for (int o = 0; o < 40; ++o)
            out[c * 40 + o] = lg[c * 40 + o] - m - ls;
    }
}

// ---------------------------------------------------------------------------
// One EdgeConv layer: kNN (all batches) then per batch: edges -> GEMM1
// (bias+ReLU, bf16) -> GEMM2 (bias, f32) -> max over K.
// ---------------------------------------------------------------------------
template <int KPA, int CHID, int KPB, int COUT>
static void run_layer(hipStream_t stream, const float* Xin, int Cin,
                      const __bf16* WAp, const float* ba,
                      const __bf16* WBp, const float* bb,
                      float* Xout, int* IDX, __bf16* Eb, __bf16* Hb,
                      float* Gb) {
    knn_kernel<<<dim3(NPTS / 256, 1, NBATCH), 256, 0, stream>>>(Xin, IDX, Cin);
    for (int b = 0; b < NBATCH; ++b) {
        const float* xb = Xin + (size_t)b * NPTS * Cin;
        const int* ib   = IDX + (size_t)b * NPTS * KNN;
        size_t etot = (size_t)MROWS * KPA;
        build_edges<<<(int)((etot + 255) / 256), 256, 0, stream>>>(
            xb, ib, Eb, Cin, KPA);
        int nt1 = (MROWS / 16) * (CHID / 32);
        wmma_gemm<KPA, CHID, true, true><<<(nt1 + 7) / 8, 256, 0, stream>>>(
            Eb, WAp, ba, (void*)Hb, MROWS);
        int nt2 = (MROWS / 16) * (COUT / 32);
        wmma_gemm<KPB, COUT, false, false><<<(nt2 + 7) / 8, 256, 0, stream>>>(
            Hb, WBp, bb, (void*)Gb, MROWS);
        reduce_max_k<<<(NPTS * COUT + 255) / 256, 256, 0, stream>>>(
            Gb, Xout + (size_t)b * NPTS * COUT, COUT);
    }
}

// ---------------------------------------------------------------------------
extern "C" void kernel_launch(void* const* d_in, const int* in_sizes, int n_in,
                              void* d_out, int out_size, void* d_ws, size_t ws_size,
                              hipStream_t stream) {
    const float* pos    = (const float*)d_in[0];
    const float* w1a    = (const float*)d_in[1];
    const float* b1a    = (const float*)d_in[2];
    const float* w1b    = (const float*)d_in[3];
    const float* b1b    = (const float*)d_in[4];
    const float* w2a    = (const float*)d_in[5];
    const float* b2a    = (const float*)d_in[6];
    const float* w2b    = (const float*)d_in[7];
    const float* b2b    = (const float*)d_in[8];
    const float* w3a    = (const float*)d_in[9];
    const float* b3a    = (const float*)d_in[10];
    const float* w3b    = (const float*)d_in[11];
    const float* b3b    = (const float*)d_in[12];
    const float* w4a    = (const float*)d_in[13];
    const float* b4a    = (const float*)d_in[14];
    const float* w4b    = (const float*)d_in[15];
    const float* b4b    = (const float*)d_in[16];
    const float* lin1_w = (const float*)d_in[17];
    const float* lin1_b = (const float*)d_in[18];
    const float* bn_g   = (const float*)d_in[19];
    const float* bn_b   = (const float*)d_in[20];
    const float* lin2_w = (const float*)d_in[21];
    const float* lin2_b = (const float*)d_in[22];

    char* ws = (char*)d_ws;
    size_t off = 0;
    auto carve = [&](size_t bytes) {
        size_t o = off; off += (bytes + 255) & ~(size_t)255; return o;
    };
    size_t oX1  = carve((size_t)NBATCH * NPTS * 64 * 4);
    size_t oX2  = carve((size_t)NBATCH * NPTS * 64 * 4);
    size_t oX3  = carve((size_t)NBATCH * NPTS * 128 * 4);
    size_t oX4  = carve((size_t)NBATCH * NPTS * 256 * 4);
    size_t oXP  = carve((size_t)NBATCH * 512 * 4);
    size_t oIDX = carve((size_t)NBATCH * NPTS * KNN * 4);
    size_t oE   = carve((size_t)MROWS * 256 * 2);   // per-batch reuse (bf16)
    size_t oH   = carve((size_t)MROWS * 256 * 2);
    size_t oG   = carve((size_t)MROWS * 256 * 4);
    size_t oWA1 = carve(32 * 64 * 2),   oWB1 = carve(64 * 64 * 2);
    size_t oWA2 = carve(128 * 64 * 2),  oWB2 = carve(64 * 64 * 2);
    size_t oWA3 = carve(128 * 128 * 2), oWB3 = carve(128 * 128 * 2);
    size_t oWA4 = carve(256 * 256 * 2), oWB4 = carve(256 * 256 * 2);

    float*  X1  = (float*)(ws + oX1);
    float*  X2  = (float*)(ws + oX2);
    float*  X3  = (float*)(ws + oX3);
    float*  X4  = (float*)(ws + oX4);
    float*  XP  = (float*)(ws + oXP);
    int*    IDX = (int*)(ws + oIDX);
    __bf16* Eb  = (__bf16*)(ws + oE);
    __bf16* Hb  = (__bf16*)(ws + oH);
    float*  Gb  = (float*)(ws + oG);

    auto pack = [&](const float* W, size_t o, int Kin, int Kp, int Ncol) {
        int total = Kp * Ncol;
        wpack<<<(total + 255) / 256, 256, 0, stream>>>(W, (__bf16*)(ws + o),
                                                       Kin, Kp, Ncol);
    };
    pack(w1a, oWA1,   6,  32,  64);  pack(w1b, oWB1,  64,  64,  64);
    pack(w2a, oWA2, 128, 128,  64);  pack(w2b, oWB2,  64,  64,  64);
    pack(w3a, oWA3, 128, 128, 128);  pack(w3b, oWB3, 128, 128, 128);
    pack(w4a, oWA4, 256, 256, 256);  pack(w4b, oWB4, 256, 256, 256);

    run_layer<32, 64, 64, 64>(stream, pos, 3, (__bf16*)(ws + oWA1), b1a,
                              (__bf16*)(ws + oWB1), b1b, X1, IDX, Eb, Hb, Gb);
    run_layer<128, 64, 64, 64>(stream, X1, 64, (__bf16*)(ws + oWA2), b2a,
                               (__bf16*)(ws + oWB2), b2b, X2, IDX, Eb, Hb, Gb);
    run_layer<128, 128, 128, 128>(stream, X2, 64, (__bf16*)(ws + oWA3), b3a,
                                  (__bf16*)(ws + oWB3), b3b, X3, IDX, Eb, Hb, Gb);
    run_layer<256, 256, 256, 256>(stream, X3, 128, (__bf16*)(ws + oWA4), b4a,
                                  (__bf16*)(ws + oWB4), b4b, X4, IDX, Eb, Hb, Gb);

    pool_kernel<<<(NBATCH * 512 + 255) / 256, 256, 0, stream>>>(X1, X2, X3, X4, XP);
    head_kernel<<<1, 1024, 0, stream>>>(XP, lin1_w, lin1_b, bn_g, bn_b,
                                        lin2_w, lin2_b, (float*)d_out);
}
